// s4_u_net_74972949119511
// MI455X (gfx1250) — matrix operand order repaired and checked by reading, not compiled, not run on hardware
//
#include <hip/hip_runtime.h>

typedef __attribute__((ext_vector_type(16))) _Float16 v16h;
typedef __attribute__((ext_vector_type(8)))  _Float16 v8h;
typedef __attribute__((ext_vector_type(8)))  float    v8f;
typedef __attribute__((ext_vector_type(4)))  unsigned int v4u;
typedef __attribute__((ext_vector_type(4)))  int v4i;
typedef __attribute__((ext_vector_type(8)))  int v8i;

#define PI_F 3.14159265358979323846f

#if defined(__has_builtin)
#if __has_builtin(__builtin_amdgcn_tensor_load_to_lds)
#define HAVE_TDM 1
#endif
#endif
#ifndef HAVE_TDM
#define HAVE_TDM 0
#endif

// ---------------- f32 -> f16 weight pre-conversion ----------------
__global__ void castf2h_kernel(const float* __restrict__ s, _Float16* __restrict__ d, int n) {
  int i = blockIdx.x * blockDim.x + threadIdx.x;
  if (i < n) d[i] = (_Float16)s[i];
}

// ---------------- encoder: h[b,c,t] = w[c,0]*x[b,t,0] + w[c,1]*x[b,t,1] + b[c] ----
__global__ void enc_kernel(const float* __restrict__ x, const float* __restrict__ w,
                           const float* __restrict__ bias, float* __restrict__ h,
                           int B, int L) {
  int idx = blockIdx.x * blockDim.x + threadIdx.x;      // over B*64*L
  int t = idx % L;
  int c = (idx / L) & 63;
  int b = idx / (L * 64);
  const float* xp = x + ((size_t)b * L + t) * 2;
  h[idx] = w[c * 2 + 0] * xp[0] + w[c * 2 + 1] * xp[1] + bias[c];
}

// ---------------- diagonal S4 scan: one wave per (b,h), one lane per state n ------
// s_t = lambda * s_{t-1} + u_t ;  y_t = 2*Re(sum_n C'_n s_t[n]) + D*u_t
__global__ void s4_scan_kernel(const float* __restrict__ u, const float* __restrict__ log_dt,
                               const float* __restrict__ C2, const float* __restrict__ Dp,
                               float* __restrict__ y, int H, int L) {
  int wid  = (blockIdx.x * blockDim.x + threadIdx.x) >> 5;  // = b*H + h
  int lane = threadIdx.x & 31;                              // state index n
  int h = wid % H;
  float dt   = __expf(log_dt[h]);
  float aRe  = -0.5f * dt, aIm = PI_F * (float)lane * dt;
  float er   = __expf(aRe);
  float lamRe = er * __cosf(aIm), lamIm = er * __sinf(aIm);
  float A_re = -0.5f, A_im = PI_F * (float)lane;
  float den  = A_re * A_re + A_im * A_im;
  float nRe  = lamRe - 1.0f, nIm = lamIm;
  float fRe  = (nRe * A_re + nIm * A_im) / den;             // (lambda-1)/A
  float fIm  = (nIm * A_re - nRe * A_im) / den;
  float cR = C2[((size_t)h * 32 + lane) * 2 + 0];
  float cI = C2[((size_t)h * 32 + lane) * 2 + 1];
  float CpRe = cR * fRe - cI * fIm;
  float CpIm = cR * fIm + cI * fRe;
  float Dh = Dp[h];
  const float* ub = u + (size_t)wid * L;
  float*       yb = y + (size_t)wid * L;
  float sRe = 0.f, sIm = 0.f;
  for (int t = 0; t < L; ++t) {
    float ut  = ub[t];
    float tRe = lamRe * sRe - lamIm * sIm + ut;
    float tIm = lamRe * sIm + lamIm * sRe;
    sRe = tRe; sIm = tIm;
    float p = CpRe * sRe - CpIm * sIm;
    p += __shfl_xor(p, 1);  p += __shfl_xor(p, 2);  p += __shfl_xor(p, 4);
    p += __shfl_xor(p, 8);  p += __shfl_xor(p, 16);
    if (lane == 0) yb[t] = 2.0f * p + Dh * ut;
  }
}

// ---------------- S4 final-timestep only (for s4c feeding 1x1 decoders) -----------
__global__ void s4_last_kernel(const float* __restrict__ u, const float* __restrict__ log_dt,
                               const float* __restrict__ C2, const float* __restrict__ Dp,
                               float* __restrict__ outlast, int H, int L) {
  int wid  = (blockIdx.x * blockDim.x + threadIdx.x) >> 5;  // = b*H + h
  int lane = threadIdx.x & 31;
  int h = wid % H;
  float dt   = __expf(log_dt[h]);
  float aRe  = -0.5f * dt, aIm = PI_F * (float)lane * dt;
  float er   = __expf(aRe);
  float lamRe = er * __cosf(aIm), lamIm = er * __sinf(aIm);
  float A_re = -0.5f, A_im = PI_F * (float)lane;
  float den  = A_re * A_re + A_im * A_im;
  float nRe  = lamRe - 1.0f, nIm = lamIm;
  float fRe  = (nRe * A_re + nIm * A_im) / den;
  float fIm  = (nIm * A_re - nRe * A_im) / den;
  float cR = C2[((size_t)h * 32 + lane) * 2 + 0];
  float cI = C2[((size_t)h * 32 + lane) * 2 + 1];
  float CpRe = cR * fRe - cI * fIm;
  float CpIm = cR * fIm + cI * fRe;
  float Dh = Dp[h];
  const float* ub = u + (size_t)wid * L;
  float sRe = 0.f, sIm = 0.f, ulast = 0.f;
  for (int t = 0; t < L; ++t) {
    float ut  = ub[t];
    float tRe = lamRe * sRe - lamIm * sIm + ut;
    float tIm = lamRe * sIm + lamIm * sRe;
    sRe = tRe; sIm = tIm; ulast = ut;
  }
  float p = CpRe * sRe - CpIm * sIm;
  p += __shfl_xor(p, 1);  p += __shfl_xor(p, 2);  p += __shfl_xor(p, 4);
  p += __shfl_xor(p, 8);  p += __shfl_xor(p, 16);
  if (lane == 0) outlast[wid] = 2.0f * p + Dh * ulast;
}

// ---------------- dilated stride-4 conv as im2col GEMM via WMMA f16->f32 ----------
// out[b, chanOff+o, t'] = bias[o] + sum_{c,k} w[o,c,k] * x[b,c,4t' + dil*k - pad]
// M = O (weights, f16, staged to LDS via TDM), K = 3C, N = B*Lo.
// Each wave owns 4 N-tiles sharing one A fragment -> 4 v_wmma per K-step.
// Also accumulates per-channel sum / sumsq (post-lrelu) for BatchNorm.
__global__ void conv_wmma_kernel(const float* __restrict__ xg, const _Float16* __restrict__ wh,
                                 const float* __restrict__ bias, float* __restrict__ out,
                                 float* __restrict__ sums,
                                 int C, int O, int L, int Lo, int B,
                                 int dil, int pad, int doLrelu, int chanOff, int OCtot) {
  __shared__ __align__(16) _Float16 smA[16 * 768];
  __shared__ int cLtab[768];
  __shared__ int offtab[768];
  const int KK  = C * 3;
  const int tid = threadIdx.x;
  const int m0  = blockIdx.y * 16;

  // im2col index tables (once per block)
  for (int kk = tid; kk < KK; kk += 256) {
    int c = kk / 3, kr = kk - 3 * c;
    cLtab[kk]  = c * L;
    offtab[kk] = dil * kr - pad;
  }

  // stage A tile (16 x KK f16 = rows m0..m0+15, contiguous in global) into LDS
#if HAVE_TDM
  if (tid < 32) {                     // one wave issues the Tensor Data Mover op
    unsigned ldsoff = (unsigned)(size_t)(&smA[0]);
    unsigned long long ga = (unsigned long long)(size_t)(wh + (size_t)m0 * KK);
    // D# group0: count=1 | lds_addr | global_addr | type=2 (bits 127:126)
    v4u g0 = {1u, ldsoff, (unsigned)ga, (unsigned)(ga >> 32) | 0x80000000u};
    // D# group1: data_size=1 (2B) | tensor_dim0=KK | tensor_dim1=16 |
    //            tile_dim0=KK | tile_dim1=16 | tensor_dim0_stride=KK
    v8i g1 = {0x10000, (KK & 0xffff) << 16, 16 << 16, KK << 16, 16, KK, 0, 0};
    v4i z4 = {0, 0, 0, 0};
#if __clang_major__ >= 23
    v8i z8 = {0, 0, 0, 0, 0, 0, 0, 0};
    __builtin_amdgcn_tensor_load_to_lds(g0, g1, z4, z4, z8, 0);
#else
    __builtin_amdgcn_tensor_load_to_lds(g0, g1, z4, z4, 0);
#endif
    __builtin_amdgcn_s_wait_tensorcnt(0);
  }
#else
  {
    const v8h* srcA = (const v8h*)(wh + (size_t)m0 * KK);
    v8h* dstA = (v8h*)smA;
    for (int i = tid; i < 2 * KK; i += 256) dstA[i] = srcA[i];  // 16*KK/8 chunks
  }
#endif
  __syncthreads();

  const int lane = tid & 31, wv = tid >> 5, hi = lane >> 4, ln = lane & 15;
  const int nt0 = blockIdx.x * 32 + wv * 4;
  int bj[4], t4[4];
  const float* xb[4];
#pragma unroll
  for (int j = 0; j < 4; ++j) {
    int n = (nt0 + j) * 16 + ln;
    int b = n / Lo, t = n - b * Lo;
    bj[j] = b; t4[j] = t * 4;
    xb[j] = xg + (size_t)b * C * L;
  }
  v8f zero = {0.f, 0.f, 0.f, 0.f, 0.f, 0.f, 0.f, 0.f};
  v8f acc[4] = {zero, zero, zero, zero};

  const _Float16* arow = smA + ln * KK + (hi << 3);
  for (int k0 = 0; k0 < KK; k0 += 32) {
    // A fragment: two contiguous 16B LDS loads per the 16-bit 16x32 ISA layout
    v8h alo = *(const v8h*)(arow + k0);
    v8h ahi = *(const v8h*)(arow + k0 + 16);
    v16h af = __builtin_shufflevector(alo, ahi, 0, 1, 2, 3, 4, 5, 6, 7,
                                      8, 9, 10, 11, 12, 13, 14, 15);
    int cLq[16], offq[16];
#pragma unroll
    for (int q = 0; q < 16; ++q) {
      int kk = k0 + (hi << 3) + (q & 7) + ((q >> 3) << 4);
      cLq[q]  = cLtab[kk];
      offq[q] = offtab[kk];
    }
    if (k0 + 32 < KK)
      __builtin_prefetch(&xb[0][cLtab[k0 + 31] + t4[0]], 0, 1);
#pragma unroll
    for (int j = 0; j < 4; ++j) {
      v16h bf;
#pragma unroll
      for (int q = 0; q < 16; ++q) {
        int pos = t4[j] + offq[q];
        float v = 0.f;
        if ((unsigned)pos < (unsigned)L) v = xb[j][cLq[q] + pos];
        bf[q] = (_Float16)v;
      }
      acc[j] = __builtin_amdgcn_wmma_f32_16x16x32_f16(false, af, false, bf,
                                                      (short)0, acc[j], false, false);
    }
  }

  // epilogue: bias, lrelu, store, BN stats (combine 4 N-tiles before reducing)
  float s1a[8] = {0.f, 0.f, 0.f, 0.f, 0.f, 0.f, 0.f, 0.f};
  float s2a[8] = {0.f, 0.f, 0.f, 0.f, 0.f, 0.f, 0.f, 0.f};
#pragma unroll
  for (int j = 0; j < 4; ++j) {
    int b = bj[j], t = t4[j] >> 2;
#pragma unroll
    for (int r = 0; r < 8; ++r) {
      int ch = m0 + r + (hi << 3);
      float v = acc[j][r] + bias[ch];
      if (doLrelu) v = (v >= 0.f) ? v : 0.3f * v;
      out[((size_t)b * OCtot + chanOff + ch) * Lo + t] = v;
      s1a[r] += v; s2a[r] += v * v;
    }
  }
#pragma unroll
  for (int r = 0; r < 8; ++r) {
    int ch = m0 + r + (hi << 3);
    float s1 = s1a[r], s2 = s2a[r];
    s1 += __shfl_xor(s1, 1);  s2 += __shfl_xor(s2, 1);
    s1 += __shfl_xor(s1, 2);  s2 += __shfl_xor(s2, 2);
    s1 += __shfl_xor(s1, 4);  s2 += __shfl_xor(s2, 4);
    s1 += __shfl_xor(s1, 8);  s2 += __shfl_xor(s2, 8);
    if (ln == 0) {
      atomicAdd(&sums[ch], s1);
      atomicAdd(&sums[O + ch], s2);
    }
  }
}

// ---------------- BatchNorm normalize (in place), stats per conv per channel ------
__global__ void bn_kernel(float* __restrict__ xio, const float* __restrict__ sums,
                          const float* __restrict__ g, const float* __restrict__ bta,
                          int Cper, int OCtot, int Lo, int B) {
  int idx = blockIdx.x * blockDim.x + threadIdx.x;      // over B*OCtot*Lo
  int ch = (idx / Lo) % OCtot;
  int conv = ch / Cper, c = ch - conv * Cper;
  const float* s = sums + conv * 2 * Cper;
  float cnt  = (float)(B * Lo);
  float mean = s[c] / cnt;
  float var  = s[Cper + c] / cnt - mean * mean;
  float v = xio[idx];
  xio[idx] = (v - mean) * rsqrtf(var + 1e-5f) * g[c] + bta[c];
}

__global__ void zero_kernel(float* __restrict__ p, int n) {
  int i = blockIdx.x * blockDim.x + threadIdx.x;
  if (i < n) p[i] = 0.f;
}

// ---------------- decoder: two 1x1 convs on the last timestep only ----------------
__global__ void dec_kernel(const float* __restrict__ x3, const float* __restrict__ w1,
                           const float* __restrict__ b1, const float* __restrict__ w2,
                           const float* __restrict__ b2, float* __restrict__ out) {
  __shared__ float sm[256];
  int b = blockIdx.x, o = threadIdx.x;
  float acc = b1[o];
  const float* xr = x3 + (size_t)b * 1024;
  const float* wr = w1 + (size_t)o * 1024;
  for (int hh = 0; hh < 1024; ++hh) acc += wr[hh] * xr[hh];
  sm[o] = acc * w2[o];
  __syncthreads();
  for (int s = 128; s > 0; s >>= 1) {
    if (o < s) sm[o] += sm[o + s];
    __syncthreads();
  }
  if (o == 0) out[b] = sm[0] + b2[0];
}

extern "C" void kernel_launch(void* const* d_in, const int* in_sizes, int n_in,
                              void* d_out, int out_size, void* d_ws, size_t ws_size,
                              hipStream_t stream) {
  (void)in_sizes; (void)n_in; (void)out_size; (void)ws_size;
  const float* x         = (const float*)d_in[0];
  const float* enc_w     = (const float*)d_in[1];
  const float* enc_b     = (const float*)d_in[2];
  const float* s4a_logdt = (const float*)d_in[3];
  const float* s4a_C     = (const float*)d_in[4];
  const float* s4a_D     = (const float*)d_in[5];
  const float* s4b_logdt = (const float*)d_in[6];
  const float* s4b_C     = (const float*)d_in[7];
  const float* s4b_D     = (const float*)d_in[8];
  const float* s4c_logdt = (const float*)d_in[9];
  const float* s4c_C     = (const float*)d_in[10];
  const float* s4c_D     = (const float*)d_in[11];
  const float* b1_w[4] = {(const float*)d_in[12], (const float*)d_in[14],
                          (const float*)d_in[16], (const float*)d_in[18]};
  const float* b1_b[4] = {(const float*)d_in[13], (const float*)d_in[15],
                          (const float*)d_in[17], (const float*)d_in[19]};
  const float* b2_w[4] = {(const float*)d_in[20], (const float*)d_in[22],
                          (const float*)d_in[24], (const float*)d_in[26]};
  const float* b2_b[4] = {(const float*)d_in[21], (const float*)d_in[23],
                          (const float*)d_in[25], (const float*)d_in[27]};
  const float* bn1_g  = (const float*)d_in[28];
  const float* bn1_b  = (const float*)d_in[29];
  const float* bn2_g  = (const float*)d_in[30];
  const float* bn2_b  = (const float*)d_in[31];
  const float* dec1_w = (const float*)d_in[32];
  const float* dec1_b = (const float*)d_in[33];
  const float* dec2_w = (const float*)d_in[34];
  const float* dec2_b = (const float*)d_in[35];

  float* ws     = (float*)d_ws;
  float* bufA   = ws;                 // 8.4M f32 (33.5 MB)
  float* bufB   = ws + 8388608;       // 8.4M f32
  float* sums   = ws + 16777216;      // up to 4*2*256 = 2048 f32
  float* x3last = sums + 4096;        // 16*1024 f32
  _Float16* whb = (_Float16*)(ws + 16777216 + 4096 + 16384);  // f16 weights
  _Float16* wh1[4], *wh2[4];
  for (int i = 0; i < 4; ++i) wh1[i] = whb + (size_t)i * (64 * 192);
  for (int i = 0; i < 4; ++i) wh2[i] = whb + 4 * (64 * 192) + (size_t)i * (256 * 768);

  const int B = 16, L0 = 8192;
  const int dils[4] = {1, 2, 4, 8};

  // pre-convert conv weights to f16 (row-major [O][3C])
  for (int i = 0; i < 4; ++i)
    castf2h_kernel<<<(64 * 192) / 256, 256, 0, stream>>>(b1_w[i], wh1[i], 64 * 192);
  for (int i = 0; i < 4; ++i)
    castf2h_kernel<<<(256 * 768) / 256, 256, 0, stream>>>(b2_w[i], wh2[i], 256 * 768);

  // encoder: [B,L,2] -> bufA [B,64,8192]
  enc_kernel<<<(B * 64 * L0) / 256, 256, 0, stream>>>(x, enc_w, enc_b, bufA, B, L0);
  // s4a: bufA -> bufB [B,64,8192]
  s4_scan_kernel<<<(B * 64) / 8, 256, 0, stream>>>(bufA, s4a_logdt, s4a_C, s4a_D, bufB, 64, L0);
  // block 1: 4 dilated convs (64->64, stride 4) -> bufA [B,256,2048] + BN stats
  zero_kernel<<<8, 256, 0, stream>>>(sums, 2048);
  for (int i = 0; i < 4; ++i) {
    dim3 g((B * 2048 / 16) / 32, 64 / 16);
    conv_wmma_kernel<<<g, 256, 0, stream>>>(bufB, wh1[i], b1_b[i], bufA, sums + i * 2 * 64,
                                            64, 64, L0, 2048, B,
                                            dils[i], dils[i], (i > 0) ? 1 : 0, i * 64, 256);
  }
  bn_kernel<<<(B * 256 * 2048) / 256, 256, 0, stream>>>(bufA, sums, bn1_g, bn1_b, 64, 256, 2048, B);
  // s4b: bufA [B,256,2048] -> bufB
  s4_scan_kernel<<<(B * 256) / 8, 256, 0, stream>>>(bufA, s4b_logdt, s4b_C, s4b_D, bufB, 256, 2048);
  // block 2: 4 dilated convs (256->256, stride 4) -> bufA [B,1024,512] + BN stats
  zero_kernel<<<8, 256, 0, stream>>>(sums, 2048);
  for (int i = 0; i < 4; ++i) {
    dim3 g((B * 512 / 16) / 32, 256 / 16);
    conv_wmma_kernel<<<g, 256, 0, stream>>>(bufB, wh2[i], b2_b[i], bufA, sums + i * 2 * 256,
                                            256, 256, 2048, 512, B,
                                            dils[i], dils[i], (i > 0) ? 1 : 0, i * 256, 1024);
  }
  bn_kernel<<<(B * 1024 * 512) / 256, 256, 0, stream>>>(bufA, sums, bn2_g, bn2_b, 256, 1024, 512, B);
  // s4c final state only -> x3last [B,1024]
  s4_last_kernel<<<(B * 1024) / 8, 256, 0, stream>>>(bufA, s4c_logdt, s4c_C, s4c_D, x3last, 1024, 512);
  // decoders on last timestep -> d_out [B]
  dec_kernel<<<16, 256, 0, stream>>>(x3last, dec1_w, dec1_b, dec2_w, dec2_b, (float*)d_out);
}
